// Attention_2147483648522
// MI455X (gfx1250) — matrix operand order repaired and checked by reading, not compiled, not run on hardware
//
#include <hip/hip_runtime.h>
#include <hip/hip_bf16.h>

typedef __attribute__((ext_vector_type(16))) __bf16 v16bf;
typedef __attribute__((ext_vector_type(8)))  float  v8f;

#define HEADS   12
#define DH      64
#define DIMM    768
#define NTOK    1025
#define NPAD    1088        // 34*32 = 17*64, K/N pad for WMMA supertiles
#define BATCH   8
#define QKVN    2304
#define MQKV    (BATCH*NTOK)    // 8200
#define MQKVPAD 8224            // 257*32
#define ROWS    257
#define HIDM    (BATCH*ROWS)    // 2056
#define HIDMPAD 2080            // 65*32
#define BHN     (BATCH*HEADS)   // 96
#define KOUT    256

// ---------- bf16 helpers (RNE) ----------
__device__ __forceinline__ unsigned short f2bf(float f) {
  unsigned int u = __float_as_uint(f);
  u += 0x7FFFu + ((u >> 16) & 1u);
  return (unsigned short)(u >> 16);
}
__device__ __forceinline__ float bf2f(unsigned short h) {
  return __uint_as_float(((unsigned int)h) << 16);
}

union FragAB { v16bf v; unsigned int u[8]; };
union FragC  { v8f  v; float f[8]; };

#define WMMA_BF16(A, B, C) \
  __builtin_amdgcn_wmma_f32_16x16x32_bf16(false, (A), false, (B), (short)0, (C), false, false)

// A-fragment: 16x32 bf16, row-major source [row][k], ld in elements.
// ISA 7.12.2: lane<16 -> M=lane, VGPR0..3 K=2i..2i+1, VGPR4..7 K=16+2i;
// lane>=16 same M, K offset +8 / +24.
__device__ __forceinline__ v16bf load_fragA(const unsigned short* base, long ld,
                                            int m0, int k0, int lane) {
  FragAB fr;
  int m = m0 + (lane & 15);
  int koff = (lane & 16) ? 8 : 0;
  const unsigned short* p = base + (long)m * ld + k0 + koff;
#pragma unroll
  for (int i = 0; i < 4; ++i) fr.u[i]     = *(const unsigned int*)(p + 2 * i);
#pragma unroll
  for (int i = 0; i < 4; ++i) fr.u[4 + i] = *(const unsigned int*)(p + 16 + 2 * i);
  return fr.v;
}

// B-fragment: 32x16 (KxN) bf16 from transposed storage BT[n][k], ld in elements.
// B striping matches A: lane<16 VGPR0..3 K=2i..2i+1, VGPR4..7 K=16+2i; lane>=16 K offset +8 / +24.
__device__ __forceinline__ v16bf load_fragB(const unsigned short* baseT, long ld,
                                            int n0, int k0, int lane) {
  FragAB fr;
  int n = n0 + (lane & 15);
  const unsigned short* p = baseT + (long)n * ld + k0 + ((lane & 16) ? 8 : 0);
#pragma unroll
  for (int i = 0; i < 4; ++i) fr.u[i]     = *(const unsigned int*)(p + 2 * i);
#pragma unroll
  for (int i = 0; i < 4; ++i) fr.u[4 + i] = *(const unsigned int*)(p + 16 + 2 * i);
  return fr.v;
}

// ---------- conversion / init ----------
__global__ void conv_x(const float* x, unsigned short* xb) {
  long i = (long)blockIdx.x * blockDim.x + threadIdx.x;
  if (i >= (long)MQKVPAD * DIMM) return;
  long m = i / DIMM;
  xb[i] = (m < MQKV) ? f2bf(x[i]) : 0;
}
__global__ void conv_wqkvT(const float* w, unsigned short* wT) {
  long i = (long)blockIdx.x * blockDim.x + threadIdx.x;
  if (i >= (long)QKVN * DIMM) return;
  long n = i / DIMM, k = i % DIMM;
  wT[i] = f2bf(w[k * QKVN + n]);
}
__global__ void conv_woT(const float* w, unsigned short* wT) {
  long i = (long)blockIdx.x * blockDim.x + threadIdx.x;
  if (i >= (long)DIMM * DIMM) return;
  long n = i / DIMM, k = i % DIMM;
  wT[i] = f2bf(w[k * DIMM + n]);
}
__global__ void zero_words(unsigned int* p, long nwords) {
  long i = (long)blockIdx.x * blockDim.x + threadIdx.x;
  if (i < nwords) p[i] = 0u;
}

// ---------- QKV GEMM: [8224x768] x [768x2304], 2Mx4N supertile per wave ----------
__global__ __launch_bounds__(32) void qkv_gemm(const unsigned short* xb,
                                               const unsigned short* wT,
                                               unsigned short* qb,
                                               unsigned short* kb,
                                               unsigned short* vtb) {
  int lane = threadIdx.x;
  int m0 = blockIdx.x * 32, n0 = blockIdx.y * 64;
  FragC c[2][4];
#pragma unroll
  for (int i = 0; i < 2; ++i)
#pragma unroll
    for (int j = 0; j < 4; ++j)
#pragma unroll
      for (int r = 0; r < 8; ++r) c[i][j].f[r] = 0.0f;

  for (int k0 = 0; k0 < DIMM; k0 += 32) {
    if (k0 + 32 < DIMM)
      __builtin_prefetch(xb + (long)(m0 + (lane & 15)) * DIMM + k0 + 32, 0, 3);
    v16bf a0 = load_fragA(xb, DIMM, m0,      k0, lane);
    v16bf a1 = load_fragA(xb, DIMM, m0 + 16, k0, lane);
#pragma unroll
    for (int j = 0; j < 4; ++j) {
      v16bf b = load_fragB(wT, DIMM, n0 + 16 * j, k0, lane);
      c[0][j].v = WMMA_BF16(a0, b, c[0][j].v);
      c[1][j].v = WMMA_BF16(a1, b, c[1][j].v);
    }
  }
  int mhi = (lane & 16) ? 8 : 0;
#pragma unroll
  for (int j = 0; j < 4; ++j) {
    int n = n0 + 16 * j + (lane & 15);
    int sec = n / DIMM, rem = n % DIMM, h = rem / DH, d = rem % DH;
#pragma unroll
    for (int mt = 0; mt < 2; ++mt)
#pragma unroll
      for (int r = 0; r < 8; ++r) {
        int m = m0 + 16 * mt + r + mhi;
        if (m >= MQKV) continue;
        int bb = m / NTOK, ni = m % NTOK;
        long bh = (long)bb * HEADS + h;
        unsigned short val = f2bf(c[mt][j].f[r]);
        if (sec == 0)      qb[(bh * NPAD + ni) * DH + d] = val;
        else if (sec == 1) kb[(bh * NPAD + ni) * DH + d] = val;
        else               vtb[(bh * DH + d) * NPAD + ni] = val;  // v transposed
      }
  }
}

// ---------- dots = q.k^T * scale, 1Mx4N supertile per wave ----------
__global__ __launch_bounds__(32) void dots_gemm(const unsigned short* qb,
                                                const unsigned short* kb,
                                                unsigned short* dots) {
  int lane = threadIdx.x;
  int m0 = blockIdx.x * 16, n0 = blockIdx.y * 64;
  long bh = blockIdx.z;
  const unsigned short* A  = qb + bh * NPAD * DH;
  const unsigned short* Bt = kb + bh * NPAD * DH;  // [token][d] == BT for B[d][token]
  FragC c[4];
#pragma unroll
  for (int j = 0; j < 4; ++j)
#pragma unroll
    for (int r = 0; r < 8; ++r) c[j].f[r] = 0.0f;
#pragma unroll
  for (int k0 = 0; k0 < DH; k0 += 32) {
    v16bf a = load_fragA(A, DH, m0, k0, lane);
#pragma unroll
    for (int j = 0; j < 4; ++j) {
      v16bf b = load_fragB(Bt, DH, n0 + 16 * j, k0, lane);
      c[j].v = WMMA_BF16(a, b, c[j].v);
    }
  }
  int mhi = (lane & 16) ? 8 : 0;
  const float scale = 0.125f;  // 64^-0.5
#pragma unroll
  for (int j = 0; j < 4; ++j) {
    int n = n0 + 16 * j + (lane & 15);
#pragma unroll
    for (int r = 0; r < 8; ++r) {
      int m = m0 + r + mhi;
      if (m < NTOK && n < NTOK)
        dots[(bh * NTOK + m) * NPAD + n] = f2bf(c[j].f[r] * scale);
    }
  }
}

// ---------- row softmax, in place, zero K-pad ----------
__global__ void softmax_rows(unsigned short* dots, const unsigned char* mask) {
  long row = blockIdx.x;
  int b = (int)(row / ((long)HEADS * NTOK));
  unsigned short* p = dots + row * NPAD;
  const unsigned char* mk = mask + (long)b * NTOK;
  __shared__ float red[128];
  int t = threadIdx.x;
  float mx = -3.4e38f;
  for (int j = t; j < NTOK; j += 128) {
    float v = mk[j] ? bf2f(p[j]) : -3.4e38f;
    mx = fmaxf(mx, v);
  }
  red[t] = mx; __syncthreads();
  for (int s = 64; s > 0; s >>= 1) {
    if (t < s) red[t] = fmaxf(red[t], red[t + s]);
    __syncthreads();
  }
  mx = red[0]; __syncthreads();
  float sum = 0.f;
  for (int j = t; j < NTOK; j += 128) {
    float v = mk[j] ? bf2f(p[j]) : -3.4e38f;
    sum += __expf(v - mx);
  }
  red[t] = sum; __syncthreads();
  for (int s = 64; s > 0; s >>= 1) {
    if (t < s) red[t] += red[t + s];
    __syncthreads();
  }
  float inv = 1.0f / red[0];
  for (int j = t; j < NPAD; j += 128) {
    if (j < NTOK) {
      float v = mk[j] ? bf2f(p[j]) : -3.4e38f;
      p[j] = f2bf(__expf(v - mx) * inv);
    } else p[j] = 0;
  }
}

// ---------- per-token value norms ----------
__global__ void val_norms(const unsigned short* vtb, float* norms) {
  long i = (long)blockIdx.x * blockDim.x + threadIdx.x;
  if (i >= (long)BHN * NTOK) return;
  long bh = i / NTOK; int n = (int)(i % NTOK);
  const unsigned short* p = vtb + bh * DH * NPAD + n;
  float s = 0.f;
#pragma unroll
  for (int d = 0; d < DH; ++d) { float v = bf2f(p[(long)d * NPAD]); s += v * v; }
  norms[i] = sqrtf(s);
}

// ---------- Threefry2x32 (JAX-compatible) ----------
__device__ __forceinline__ unsigned int rotl32(unsigned int x, int r) {
  return (x << r) | (x >> (32 - r));
}
__device__ __forceinline__ void threefry2x32(unsigned int k0, unsigned int k1,
                                             unsigned int x0, unsigned int x1,
                                             unsigned int* o0, unsigned int* o1) {
  unsigned int ks[3] = {k0, k1, 0x1BD11BDAu ^ k0 ^ k1};
  const int R1[4] = {13, 15, 26, 6}, R2[4] = {17, 29, 16, 24};
  x0 += ks[0]; x1 += ks[1];
#pragma unroll
  for (int g = 0; g < 5; ++g) {
    const int* r = (g & 1) ? R2 : R1;
#pragma unroll
    for (int i = 0; i < 4; ++i) { x0 += x1; x1 = rotl32(x1, r[i]); x1 ^= x0; }
    x0 += ks[(g + 1) % 3];
    x1 += ks[(g + 2) % 3] + (unsigned int)(g + 1);
  }
  *o0 = x0; *o1 = x1;
}

// ---------- CLS pooling + Gumbel top-k sampling + unique-sort-pad ----------
__global__ void sample_tokens(const unsigned short* attn, const float* norms,
                              const unsigned char* mask, int* uidsI,
                              float* outF, long offMask, long offUid) {
  int b = blockIdx.x;          // one block per batch, 256 threads
  int t = threadIdx.x;
  __shared__ float plog[NTOK - 1];
  __shared__ float red[256];
  __shared__ unsigned int present[33];
  if (t < 33) present[t] = 0u;

  float part = 0.f;
  for (int n = t; n < NTOK - 1; n += 256) {
    float cls = 0.f;
#pragma unroll
    for (int h = 0; h < HEADS; ++h) {
      long bh = (long)b * HEADS + h;
      float a  = bf2f(attn[(bh * NTOK + 0) * NPAD + 1 + n]);
      float vn = norms[bh * NTOK + 1 + n];
      cls += a * vn;
    }
    plog[n] = cls;
    part += cls;
  }
  red[t] = part; __syncthreads();
  for (int s = 128; s > 0; s >>= 1) {
    if (t < s) red[t] += red[t + s];
    __syncthreads();
  }
  float total = red[0] + 1e-6f;
  for (int n = t; n < NTOK - 1; n += 256) {
    float pl = logf(plog[n] / total + 1e-6f);
    if (!mask[(long)b * NTOK + 1 + n]) pl = -1.7e38f;  // -big/2
    plog[n] = pl;
  }
  __syncthreads();

  // each thread: one Gumbel-argmax sample (u = jax.random.uniform(key(42)))
  long base = ((long)b * KOUT + t) * (NTOK - 1);
  const unsigned int HALF = 1u << 20;  // total count 2^21, split halves
  float best = -3.4e38f; int bestn = 0;
  for (int n = 0; n < NTOK - 1; ++n) {
    unsigned int idx = (unsigned int)(base + n);
    unsigned int x0, x1, o0, o1;
    if (idx < HALF) { x0 = idx; x1 = idx + HALF; } else { x0 = idx - HALF; x1 = idx; }
    threefry2x32(0u, 42u, x0, x1, &o0, &o1);
    unsigned int bits = (idx < HALF) ? o0 : o1;
    float u = fmaxf(0.f, __uint_as_float((bits >> 9) | 0x3F800000u) - 1.0f);
    float g = -logf(-logf(u + 1e-6f) + 1e-6f);
    float v = plog[n] + g;
    if (v > best) { best = v; bestn = n; }
  }
  int sampled = bestn + 1;  // 1..1024
  atomicOr(&present[sampled >> 5], 1u << (sampled & 31));
  __syncthreads();

  if (t == 0) {  // dedup+sort+pad == compact ascending from bitmap
    int cnt = 1;
    uidsI[b * ROWS + 0] = 0;
    for (int id = 1; id <= 1024; ++id)
      if ((present[id >> 5] >> (id & 31)) & 1u) uidsI[b * ROWS + cnt++] = id;
    for (; cnt < ROWS; ++cnt) uidsI[b * ROWS + cnt] = 0;
    for (int i = 0; i < ROWS; ++i) {
      int id = uidsI[b * ROWS + i];
      outF[offMask + (long)b * ROWS + i] = (i == 0 || id != 0) ? 1.0f : 0.0f;
      outF[offUid  + (long)b * ROWS + i] = (float)id;
    }
  }
}

// ---------- gathered attn @ v, 1Mx4N (full DH=64) per wave ----------
__global__ __launch_bounds__(32) void attnv_gemm(const unsigned short* attn,
                                                 const unsigned short* vtb,
                                                 const int* uidsI,
                                                 unsigned short* hid) {
  int lane = threadIdx.x;
  int m0 = blockIdx.x * 16;
  long bh = blockIdx.y;
  int b = (int)(bh / HEADS), h = (int)(bh % HEADS);
  FragC c[4];
#pragma unroll
  for (int j = 0; j < 4; ++j)
#pragma unroll
    for (int r = 0; r < 8; ++r) c[j].f[r] = 0.0f;

  int mi = m0 + (lane & 15);
  int koff = (lane & 16) ? 8 : 0;
  int rowid = (mi < ROWS) ? uidsI[b * ROWS + mi] : -1;
  const unsigned short* vt = vtb + bh * DH * NPAD;
  for (int k0 = 0; k0 < NPAD; k0 += 32) {
    FragAB a;
    if (rowid >= 0) {
      const unsigned short* p = attn + (bh * NTOK + rowid) * NPAD + k0 + koff;
#pragma unroll
      for (int i = 0; i < 4; ++i) a.u[i]     = *(const unsigned int*)(p + 2 * i);
#pragma unroll
      for (int i = 0; i < 4; ++i) a.u[4 + i] = *(const unsigned int*)(p + 16 + 2 * i);
    } else {
#pragma unroll
      for (int i = 0; i < 8; ++i) a.u[i] = 0u;
    }
#pragma unroll
    for (int j = 0; j < 4; ++j) {
      v16bf bf = load_fragB(vt, NPAD, 16 * j, k0, lane);  // BT[d][j]
      c[j].v = WMMA_BF16(a.v, bf, c[j].v);
    }
  }
  int mhi = (lane & 16) ? 8 : 0;
#pragma unroll
  for (int j = 0; j < 4; ++j) {
    int n = 16 * j + (lane & 15);  // d index within head
#pragma unroll
    for (int r = 0; r < 8; ++r) {
      int m = m0 + r + mhi;
      if (m < ROWS)
        hid[((long)(b * ROWS + m)) * DIMM + h * DH + n] = f2bf(c[j].f[r]);
    }
  }
}

// ---------- out = hidden @ w_out + b_out, 2Mx4N supertile per wave ----------
__global__ __launch_bounds__(32) void out_gemm(const unsigned short* hid,
                                               const unsigned short* woT,
                                               const float* bias, float* out) {
  int lane = threadIdx.x;
  int m0 = blockIdx.x * 32, n0 = blockIdx.y * 64;
  FragC c[2][4];
#pragma unroll
  for (int i = 0; i < 2; ++i)
#pragma unroll
    for (int j = 0; j < 4; ++j)
#pragma unroll
      for (int r = 0; r < 8; ++r) c[i][j].f[r] = 0.0f;

  for (int k0 = 0; k0 < DIMM; k0 += 32) {
    if (k0 + 32 < DIMM)
      __builtin_prefetch(hid + (long)(m0 + (lane & 15)) * DIMM + k0 + 32, 0, 3);
    v16bf a0 = load_fragA(hid, DIMM, m0,      k0, lane);
    v16bf a1 = load_fragA(hid, DIMM, m0 + 16, k0, lane);
#pragma unroll
    for (int j = 0; j < 4; ++j) {
      v16bf b = load_fragB(woT, DIMM, n0 + 16 * j, k0, lane);
      c[0][j].v = WMMA_BF16(a0, b, c[0][j].v);
      c[1][j].v = WMMA_BF16(a1, b, c[1][j].v);
    }
  }
  int mhi = (lane & 16) ? 8 : 0;
#pragma unroll
  for (int j = 0; j < 4; ++j) {
    int n = n0 + 16 * j + (lane & 15);
    float bs = bias[n];
#pragma unroll
    for (int mt = 0; mt < 2; ++mt)
#pragma unroll
      for (int r = 0; r < 8; ++r) {
        int m = m0 + 16 * mt + r + mhi;
        if (m < HIDM) out[(long)m * DIMM + n] = c[mt][j].f[r] + bs;
      }
  }
}

extern "C" void kernel_launch(void* const* d_in, const int* in_sizes, int n_in,
                              void* d_out, int out_size, void* d_ws, size_t ws_size,
                              hipStream_t stream) {
  const float* x          = (const float*)d_in[0];
  const unsigned char* mk = (const unsigned char*)d_in[1];  // jnp bool = 1 byte
  const float* w_qkv      = (const float*)d_in[2];
  const float* w_out      = (const float*)d_in[3];
  const float* b_out      = (const float*)d_in[4];
  float* out              = (float*)d_out;

  // workspace layout
  char* ws = (char*)d_ws;
  const size_t SZ_XB  = (size_t)MQKVPAD * DIMM * 2;
  const size_t SZ_WQT = (size_t)QKVN * DIMM * 2;
  const size_t SZ_WOT = (size_t)DIMM * DIMM * 2;
  const size_t SZ_QKV = (size_t)BHN * NPAD * DH * 2;   // per q / k / vT
  const size_t SZ_HID = (size_t)HIDMPAD * DIMM * 2;
  const size_t SZ_ATT = (size_t)BHN * NTOK * NPAD * 2;
  const size_t SZ_NRM = (size_t)BHN * NTOK * 4;

  unsigned short* xb  = (unsigned short*)(ws);
  unsigned short* wqT = (unsigned short*)(ws + SZ_XB);
  unsigned short* woT = (unsigned short*)(ws + SZ_XB + SZ_WQT);
  unsigned short* qb  = (unsigned short*)(ws + SZ_XB + SZ_WQT + SZ_WOT);
  unsigned short* kb  = (unsigned short*)((char*)qb + SZ_QKV);
  unsigned short* vtb = (unsigned short*)((char*)kb + SZ_QKV);
  unsigned short* hid = (unsigned short*)((char*)vtb + SZ_QKV);
  unsigned short* att = (unsigned short*)((char*)hid + SZ_HID);
  float* norms        = (float*)((char*)att + SZ_ATT);
  int* uidsI          = (int*)((char*)norms + SZ_NRM);

  const long OUT_MAIN = (long)HIDM * DIMM;          // 1,579,008 floats
  const long OFF_MASK = OUT_MAIN;
  const long OFF_UID  = OUT_MAIN + (long)BATCH * ROWS;

  // 1) conversions + zero pad regions
  {
    long n = (long)MQKVPAD * DIMM;
    conv_x<<<dim3((unsigned)((n + 255) / 256)), 256, 0, stream>>>(x, xb);
  }
  {
    long n = (long)QKVN * DIMM;
    conv_wqkvT<<<dim3((unsigned)((n + 255) / 256)), 256, 0, stream>>>(w_qkv, wqT);
  }
  {
    long n = (long)DIMM * DIMM;
    conv_woT<<<dim3((unsigned)((n + 255) / 256)), 256, 0, stream>>>(w_out, woT);
  }
  {
    long nwords = (long)(3 * SZ_QKV + SZ_HID) / 4;   // q, k, vT, hid
    zero_words<<<dim3((unsigned)((nwords + 255) / 256)), 256, 0, stream>>>(
        (unsigned int*)qb, nwords);
  }
  // 2) QKV GEMM (2x4 supertiles)
  qkv_gemm<<<dim3(MQKVPAD / 32, QKVN / 64), 32, 0, stream>>>(xb, wqT, qb, kb, vtb);
  // 3) dots (1x4 supertiles) + softmax
  dots_gemm<<<dim3(65, NPAD / 64, BHN), 32, 0, stream>>>(qb, kb, att);
  softmax_rows<<<dim3((unsigned)((long)BHN * NTOK)), 128, 0, stream>>>(att, mk);
  // 4) norms + sampling
  {
    long n = (long)BHN * NTOK;
    val_norms<<<dim3((unsigned)((n + 255) / 256)), 256, 0, stream>>>(vtb, norms);
  }
  sample_tokens<<<dim3(BATCH), 256, 0, stream>>>(att, norms, mk, uidsI, out,
                                                 OFF_MASK, OFF_UID);
  // 5) gathered attn@v (17 M-tiles x full 64-col head), then out GEMM
  attnv_gemm<<<dim3(17, BHN), 32, 0, stream>>>(att, vtb, uidsI, hid);
  out_gemm<<<dim3(HIDMPAD / 32, DIMM / 64), 32, 0, stream>>>(hid, woT, b_out, out);
}
